// Qwen35GatedDeltaNet_77592879170050
// MI455X (gfx1250) — matrix operand order repaired and checked by reading, not compiled, not run on hardware
//
#include <hip/hip_runtime.h>
#include <hip/hip_bf16.h>

typedef __attribute__((ext_vector_type(8)))  float  v8f;
typedef __attribute__((ext_vector_type(8)))  __bf16 v8bf;
typedef __attribute__((ext_vector_type(16))) __bf16 v16bf;
typedef __attribute__((ext_vector_type(4)))  int    v4i;

#define HID      2048
#define HK       8
#define HV       16
#define DK       128
#define DV       128
#define KCONV    4
#define KEY_DIM  1024
#define VAL_DIM  2048
#define CONV_DIM 4096
#define BB       2
#define SS       1024
#define NTOK     (BB * SS)

// ---------------- async global->LDS copy (gfx1250), guarded by __has_builtin
#if __has_builtin(__builtin_amdgcn_global_load_async_to_lds_b128)
#define HAS_ASYNC_LDS 1
typedef __attribute__((address_space(1))) v4i gas_v4i;   // global int4
typedef __attribute__((address_space(3))) v4i las_v4i;   // LDS int4
__device__ __forceinline__ void async_cp16(const void* g, void* l) {
    __builtin_amdgcn_global_load_async_to_lds_b128(
        (gas_v4i*)g, (las_v4i*)l, 0, 0);
}
#if __has_builtin(__builtin_amdgcn_s_wait_asynccnt)
#define WAIT_ASYNC(n) __builtin_amdgcn_s_wait_asynccnt(n)
#else
#define WAIT_ASYNC(n) asm volatile("s_wait_asynccnt %0" ::"i"(n) : "memory")
#endif
#else
#define HAS_ASYNC_LDS 0
#endif

// ---------------------------------------------------------------- f32 -> bf16
__device__ __forceinline__ __bf16 f32_to_bf16_rne(float x) {
    unsigned u = __float_as_uint(x);
    u += 0x7FFFu + ((u >> 16) & 1u);
    unsigned short h = (unsigned short)(u >> 16);
    return __builtin_bit_cast(__bf16, h);
}

__global__ void cvt_f32_bf16(const float* __restrict__ src,
                             __bf16* __restrict__ dst, int n) {
    int i = blockIdx.x * blockDim.x + threadIdx.x;
    if (i < n) dst[i] = f32_to_bf16_rne(src[i]);
}

// ------------------------------------------------- bf16 WMMA GEMM: C = A*B^T + bias
// A: [M,K] bf16 row-major (activations), Bm: [N,K] bf16 row-major (weights),
// C: [M,N] f32.  Block = 256 thr = 8 waves; block tile 128x128; wave tile 32x64.
// A/B K-slabs (128x64 bf16) double-buffered in LDS via async copies; row stride
// padded to 72 (144 B = 36 banks -> conflict-free b128 fragment reads).
#define KSTEP 64
#define LDT   72
#define TILE_ELEMS (128 * LDT)

__global__ __launch_bounds__(256) void gemm_bf16_wmma(
    const __bf16* __restrict__ A, const __bf16* __restrict__ Bm,
    const float* __restrict__ bias, float* __restrict__ C,
    int M, int N, int K)
{
    __shared__ __align__(16) __bf16 sA[2][TILE_ELEMS];
    __shared__ __align__(16) __bf16 sB[2][TILE_ELEMS];

    const int tid  = threadIdx.x;
    const int lane = tid & 31, wave = tid >> 5;
    const int lh = lane >> 4, ln = lane & 15;
    const int mBase = blockIdx.y * 128;
    const int nBase = blockIdx.x * 128;

    v8f acc[2][4];
#pragma unroll
    for (int i = 0; i < 2; ++i)
#pragma unroll
        for (int j = 0; j < 4; ++j) acc[i][j] = (v8f)(0.f);

    // WMMA fragment reads from LDS (layouts per ISA 7.12.2); 2 k-substeps of 32
    auto compute = [&](int buf) {
#pragma unroll
        for (int kk = 0; kk < KSTEP; kk += 32) {
            union { v16bf v; v8bf h[2]; } af[2], bfr[4];
#pragma unroll
            for (int i = 0; i < 2; ++i) {
                const __bf16* p =
                    &sA[buf][((wave >> 1) * 32 + i * 16 + ln) * LDT + kk + lh * 8];
                af[i].h[0] = *(const v8bf*)p;
                af[i].h[1] = *(const v8bf*)(p + 16);
            }
#pragma unroll
            for (int j = 0; j < 4; ++j) {
                const __bf16* p =
                    &sB[buf][((wave & 1) * 64 + j * 16 + ln) * LDT + kk + lh * 8];
                bfr[j].h[0] = *(const v8bf*)p;
                bfr[j].h[1] = *(const v8bf*)(p + 16);
            }
#pragma unroll
            for (int i = 0; i < 2; ++i)
#pragma unroll
                for (int j = 0; j < 4; ++j)
                    acc[i][j] = __builtin_amdgcn_wmma_f32_16x16x32_bf16(
                        false, af[i].v, false, bfr[j].v, (short)0, acc[i][j],
                        false, false);
        }
    };

#if HAS_ASYNC_LDS
    // async engine fills LDS; 8 async b128 ops per thread per tile pair
    auto fill = [&](int buf, int k0) {
#pragma unroll
        for (int i = 0; i < 4; ++i) {
            const int seg = tid + i * 256;        // 0..1023
            const int row = seg >> 3;
            const int s16 = (seg & 7) * 8;
            async_cp16(A + (size_t)(mBase + row) * K + k0 + s16,
                       &sA[buf][row * LDT + s16]);
            async_cp16(Bm + (size_t)(nBase + row) * K + k0 + s16,
                       &sB[buf][row * LDT + s16]);
        }
    };
    fill(0, 0);
    int cur = 0;
    for (int k0 = 0; k0 < K; k0 += KSTEP) {
        __syncthreads();                       // everyone done reading buf cur^1
        const bool more = (k0 + KSTEP) < K;
        if (more) fill(cur ^ 1, k0 + KSTEP);
        if (more) { WAIT_ASYNC(8); } else { WAIT_ASYNC(0); }  // cur's 8 done (in-order)
        __syncthreads();                       // fills visible to all waves
        compute(cur);
        cur ^= 1;
    }
#else
    // fallback: regs -> ds_store double buffer
    v8bf ra[4], rb[4];
    auto gload = [&](int k0) {
#pragma unroll
        for (int i = 0; i < 4; ++i) {
            const int seg = tid + i * 256;
            const int row = seg >> 3;
            const int s16 = (seg & 7) * 8;
            ra[i] = *(const v8bf*)(A + (size_t)(mBase + row) * K + k0 + s16);
            rb[i] = *(const v8bf*)(Bm + (size_t)(nBase + row) * K + k0 + s16);
        }
    };
    auto sstore = [&](int buf) {
#pragma unroll
        for (int i = 0; i < 4; ++i) {
            const int seg = tid + i * 256;
            const int row = seg >> 3;
            const int s16 = (seg & 7) * 8;
            *(v8bf*)&sA[buf][row * LDT + s16] = ra[i];
            *(v8bf*)&sB[buf][row * LDT + s16] = rb[i];
        }
    };
    gload(0);
    sstore(0);
    int cur = 0;
    for (int k0 = 0; k0 < K; k0 += KSTEP) {
        __syncthreads();                       // tile cur ready / prev reads done
        const bool more = (k0 + KSTEP) < K;
        if (more) gload(k0 + KSTEP);           // overlap with compute
        compute(cur);
        __syncthreads();                       // reads of cur^1 (prev iter) done
        if (more) sstore(cur ^ 1);
        cur ^= 1;
    }
#endif

#pragma unroll
    for (int j = 0; j < 4; ++j) {
        const int n  = nBase + (wave & 1) * 64 + j * 16 + ln;
        const float bn = bias ? bias[n] : 0.f;
#pragma unroll
        for (int i = 0; i < 2; ++i) {
#pragma unroll
            for (int r = 0; r < 8; ++r) {
                const int m = mBase + (wave >> 1) * 32 + i * 16 + r + lh * 8;
                C[(size_t)m * N + n] = acc[i][j][r] + bn;
            }
        }
    }
}

// ---------------------------------------------- beta / g projections (N=16 each)
__global__ __launch_bounds__(256) void proj_beta_g(
    const float* __restrict__ hs, const float* __restrict__ Wb,
    const float* __restrict__ bb, const float* __restrict__ Wa,
    const float* __restrict__ ba, const float* __restrict__ dtb,
    const float* __restrict__ Alog, float* __restrict__ beta,
    float* __restrict__ g)
{
    __shared__ float row[HID];
    const int tok = blockIdx.x;
    for (int i = threadIdx.x; i < HID; i += 256) row[i] = hs[(size_t)tok * HID + i];
    __syncthreads();
    const int wave = threadIdx.x >> 5, lane = threadIdx.x & 31;
#pragma unroll
    for (int q = 0; q < 4; ++q) {
        const int d = wave * 4 + q;                       // 0..15 beta, 16..31 g
        const float* W = (d < HV) ? (Wb + (size_t)d * HID)
                                  : (Wa + (size_t)(d - HV) * HID);
        float p = 0.f;
        for (int k = lane; k < HID; k += 32) p = fmaf(row[k], W[k], p);
#pragma unroll
        for (int off = 16; off; off >>= 1) p += __shfl_xor(p, off);
        if (lane == 0) {
            if (d < HV) {
                beta[tok * HV + d] = 1.f / (1.f + expf(-(p + bb[d])));
            } else {
                const int h = d - HV;
                const float x  = p + ba[h] + dtb[h];
                const float sp = (x > 20.f) ? x : log1pf(expf(x));
                g[tok * HV + h] = -expf(Alog[h]) * sp;
            }
        }
    }
}

// ----------------------------------------- causal depthwise conv (K=4) + SiLU
__global__ void conv_silu(const float* __restrict__ mixed,
                          const float* __restrict__ cw,
                          const float* __restrict__ cb,
                          float* __restrict__ out)
{
    const int idx = blockIdx.x * blockDim.x + threadIdx.x;
    if (idx >= NTOK * CONV_DIM) return;
    const int c = idx & (CONV_DIM - 1);
    const int t = idx >> 12;           // token = b*S + s
    const int s = t & (SS - 1);
    float acc = cb[c];
#pragma unroll
    for (int j = 0; j < KCONV; ++j) {
        const int sp = s - (KCONV - 1) + j;
        if (sp >= 0)
            acc = fmaf(mixed[(size_t)(t - (KCONV - 1) + j) * CONV_DIM + c],
                       cw[c * KCONV + j], acc);
    }
    out[idx] = acc / (1.f + expf(-acc));   // SiLU
}

// -------------------------------- gated delta-rule recurrence (register state)
// One block per (b,h); 128 threads; thread v owns state column state[:,v] in VGPRs.
// k/q broadcasts read from LDS as float4 (ds_load_b128, single-bank broadcast).
__global__ __launch_bounds__(128) void delta_scan(
    const float* __restrict__ conv, const float* __restrict__ g,
    const float* __restrict__ beta, float* __restrict__ core)
{
    const int bh = blockIdx.x;       // b*HV + h
    const int b  = bh >> 4, h = bh & 15;
    const int hk = h >> 1;           // GQA repeat = 2
    const int t  = threadIdx.x;      // value column v
    const int lane = t & 31, wave = t >> 5;
    __shared__ __align__(16) float ksh[DK];
    __shared__ __align__(16) float qsh[DK];
    __shared__ float red[2][4];

    float st[DK];
#pragma unroll
    for (int i = 0; i < DK; ++i) st[i] = 0.f;
    const float qscale = 0.08838834764831845f;  // DK^-0.5

    for (int s = 0; s < SS; ++s) {
        const size_t base = (size_t)(b * SS + s) * CONV_DIM;
        const float qraw = conv[base + hk * DK + t];
        const float kraw = conv[base + KEY_DIM + hk * DK + t];
        const float vt   = conv[base + 2 * KEY_DIM + h * DV + t];

        float sq = qraw * qraw, sk = kraw * kraw;
#pragma unroll
        for (int off = 16; off; off >>= 1) {
            sq += __shfl_xor(sq, off);
            sk += __shfl_xor(sk, off);
        }
        __syncthreads();                       // protect prev-iter ksh/red reads
        if (lane == 0) { red[0][wave] = sq; red[1][wave] = sk; }
        __syncthreads();
        const float sqs = red[0][0] + red[0][1] + red[0][2] + red[0][3];
        const float sks = red[1][0] + red[1][1] + red[1][2] + red[1][3];
        qsh[t] = qraw * rsqrtf(sqs + 1e-6f) * qscale;
        ksh[t] = kraw * rsqrtf(sks + 1e-6f);
        __syncthreads();

        const float gexp = expf(g[(b * SS + s) * HV + h]);
        const float bta  = beta[(b * SS + s) * HV + h];

        const float4* k4 = (const float4*)ksh;
        const float4* q4 = (const float4*)qsh;

        float kv = 0.f;
#pragma unroll
        for (int c4 = 0; c4 < DK / 4; ++c4) {
            const float4 kf = k4[c4];
            st[4*c4+0] *= gexp; kv = fmaf(kf.x, st[4*c4+0], kv);
            st[4*c4+1] *= gexp; kv = fmaf(kf.y, st[4*c4+1], kv);
            st[4*c4+2] *= gexp; kv = fmaf(kf.z, st[4*c4+2], kv);
            st[4*c4+3] *= gexp; kv = fmaf(kf.w, st[4*c4+3], kv);
        }
        const float delta = (vt - kv) * bta;
        float o = 0.f;
#pragma unroll
        for (int c4 = 0; c4 < DK / 4; ++c4) {
            const float4 kf = k4[c4];
            const float4 qf = q4[c4];
            st[4*c4+0] = fmaf(kf.x, delta, st[4*c4+0]); o = fmaf(qf.x, st[4*c4+0], o);
            st[4*c4+1] = fmaf(kf.y, delta, st[4*c4+1]); o = fmaf(qf.y, st[4*c4+1], o);
            st[4*c4+2] = fmaf(kf.z, delta, st[4*c4+2]); o = fmaf(qf.z, st[4*c4+2], o);
            st[4*c4+3] = fmaf(kf.w, delta, st[4*c4+3]); o = fmaf(qf.w, st[4*c4+3], o);
        }
        core[((size_t)(b * SS + s) * HV + h) * DV + t] = o;
    }
}

// ------------------------- gated RMSNorm * SiLU(z), emit bf16 for output GEMM
__global__ __launch_bounds__(128) void gated_rmsnorm(
    const float* __restrict__ core, const float* __restrict__ z,
    const float* __restrict__ nw, __bf16* __restrict__ out)
{
    const int tok = blockIdx.x >> 4, h = blockIdx.x & 15;
    const int v = threadIdx.x, lane = v & 31, wave = v >> 5;
    __shared__ float red[4];
    const float c = core[((size_t)tok * HV + h) * DV + v];
    float sq = c * c;
#pragma unroll
    for (int off = 16; off; off >>= 1) sq += __shfl_xor(sq, off);
    if (lane == 0) red[wave] = sq;
    __syncthreads();
    const float var = (red[0] + red[1] + red[2] + red[3]) * (1.f / DV);
    const float zv  = z[(size_t)tok * VAL_DIM + h * DV + v];
    const float y   = c * rsqrtf(var + 1e-6f) * nw[v] * (zv / (1.f + expf(-zv)));
    out[(size_t)tok * VAL_DIM + h * DV + v] = f32_to_bf16_rne(y);
}

// ---------------------------------------------------------------- launcher
extern "C" void kernel_launch(void* const* d_in, const int* in_sizes, int n_in,
                              void* d_out, int out_size, void* d_ws, size_t ws_size,
                              hipStream_t stream)
{
    const float* hs   = (const float*)d_in[0];
    const float* Wqkv = (const float*)d_in[1];
    const float* bqkv = (const float*)d_in[2];
    const float* Wz   = (const float*)d_in[3];
    const float* bz   = (const float*)d_in[4];
    const float* Wb   = (const float*)d_in[5];
    const float* bb   = (const float*)d_in[6];
    const float* Wa   = (const float*)d_in[7];
    const float* ba   = (const float*)d_in[8];
    const float* cw   = (const float*)d_in[9];
    const float* cb   = (const float*)d_in[10];
    const float* dtb  = (const float*)d_in[11];
    const float* Alog = (const float*)d_in[12];
    const float* nw   = (const float*)d_in[13];
    const float* Wout = (const float*)d_in[14];
    const float* bout = (const float*)d_in[15];

    char* w = (char*)d_ws;
    __bf16* hsb   = (__bf16*)(w + 0);            // 2048*2048 bf16
    __bf16* Wqkvb = (__bf16*)(w + 8388608);      // 4096*2048 bf16
    __bf16* Wzb   = (__bf16*)(w + 25165824);     // 2048*2048 bf16
    __bf16* Woutb = (__bf16*)(w + 33554432);     // 2048*2048 bf16
    float*  mixed = (float*) (w + 41943040);     // 2048*4096 f32
    float*  convo = (float*) (w + 75497472);     // 2048*4096 f32
    float*  zbuf  = (float*) (w + 109051904);    // 2048*2048 f32
    float*  betab = (float*) (w + 125829120);    // 2048*16 f32
    float*  gbuf  = (float*) (w + 125960192);    // 2048*16 f32
    float*  coref = (float*) (w + 126091264);    // 2048*2048 f32
    __bf16* coreb = (__bf16*)(w + 142868480);    // 2048*2048 bf16

    // 1) down-convert activations + GEMM weights to bf16
    {
        int n;
        n = NTOK * HID;
        cvt_f32_bf16<<<(n + 255) / 256, 256, 0, stream>>>(hs, hsb, n);
        n = CONV_DIM * HID;
        cvt_f32_bf16<<<(n + 255) / 256, 256, 0, stream>>>(Wqkv, Wqkvb, n);
        n = VAL_DIM * HID;
        cvt_f32_bf16<<<(n + 255) / 256, 256, 0, stream>>>(Wz, Wzb, n);
        n = HID * VAL_DIM;
        cvt_f32_bf16<<<(n + 255) / 256, 256, 0, stream>>>(Wout, Woutb, n);
    }

    // 2) small projections: beta = sigmoid(hs@Wb.T+bb), g = -exp(A)*softplus(a+dt)
    proj_beta_g<<<NTOK, 256, 0, stream>>>(hs, Wb, bb, Wa, ba, dtb, Alog,
                                          betab, gbuf);

    // 3) big WMMA GEMMs: mixed = hs@Wqkv.T + bqkv ; z = hs@Wz.T + bz
    gemm_bf16_wmma<<<dim3(CONV_DIM / 128, NTOK / 128), 256, 0, stream>>>(
        hsb, Wqkvb, bqkv, mixed, NTOK, CONV_DIM, HID);
    gemm_bf16_wmma<<<dim3(VAL_DIM / 128, NTOK / 128), 256, 0, stream>>>(
        hsb, Wzb, bz, zbuf, NTOK, VAL_DIM, HID);

    // 4) causal depthwise conv + SiLU
    conv_silu<<<(NTOK * CONV_DIM) / 256, 256, 0, stream>>>(mixed, cw, cb, convo);

    // 5) gated delta-rule recurrence (includes q/k l2norm + scaling)
    delta_scan<<<BB * HV, 128, 0, stream>>>(convo, gbuf, betab, coref);

    // 6) gated RMSNorm * SiLU(z)  -> bf16
    gated_rmsnorm<<<NTOK * HV, 128, 0, stream>>>(coref, zbuf, nw, coreb);

    // 7) output WMMA GEMM: out = core@Wout.T + bout
    gemm_bf16_wmma<<<dim3(HID / 128, NTOK / 128), 256, 0, stream>>>(
        coreb, Woutb, bout, (float*)d_out, NTOK, HID, VAL_DIM);
}
